// Block_39058432589978
// MI455X (gfx1250) — matrix operand order repaired and checked by reading, not compiled, not run on hardware
//
#include <hip/hip_runtime.h>
#include <math.h>

typedef __bf16 bf16;
typedef __attribute__((ext_vector_type(8)))  bf16  v8bf;
typedef __attribute__((ext_vector_type(16))) bf16  v16bf;
typedef __attribute__((ext_vector_type(8)))  float v8f;
typedef __attribute__((ext_vector_type(4)))  unsigned v4u;
typedef __attribute__((ext_vector_type(8)))  int      v8i;
typedef __attribute__((ext_vector_type(4)))  int      v4i;

#define B_   2
#define T_   2048
#define C_   1024
#define H_   16
#define KV_  4
#define HD_  64
#define F_   1024
#define E_   8
#define N_   (B_*T_)
#define CKV_ (C_ + 2*KV_*HD_)   /* 1536 */

// ---------------- WMMA helpers (CDNA5 bf16 16x16x32) ----------------
static __device__ __forceinline__ v8f zero8() {
  v8f z;
#pragma unroll
  for (int i = 0; i < 8; i++) z[i] = 0.f;
  return z;
}
static __device__ __forceinline__ v8f wmma_bf(v16bf a, v16bf b, v8f c) {
  return __builtin_amdgcn_wmma_f32_16x16x32_bf16(false, a, false, b, (short)0, c, false, false);
}
static __device__ __forceinline__ v16bf cat8(v8bf lo, v8bf hi) {
  return __builtin_shufflevector(lo, hi, 0,1,2,3,4,5,6,7,8,9,10,11,12,13,14,15);
}
// A-matrix 16x32 bf16 fragment: lane<16 holds K {0..7,16..23}, lane>=16 holds K {8..15,24..31}
static __device__ __forceinline__ v16bf load_a_frag(const bf16* rowp, int kk, int lane) {
  int hi = (lane & 16) ? 8 : 0;
  v8bf lo = *(const v8bf*)(rowp + kk + hi);
  v8bf hh = *(const v8bf*)(rowp + kk + 16 + hi);
  return cat8(lo, hh);
}
// B-matrix 32x16 bf16 fragment: lane<16 holds K {0..7,16..23}, lane>=16 holds K {8..15,24..31}
static __device__ __forceinline__ v16bf load_b_frag(const bf16* rowp, int kk, int lane) {
  int hi = (lane & 16) ? 8 : 0;
  v8bf lo = *(const v8bf*)(rowp + kk + hi);
  v8bf hh = *(const v8bf*)(rowp + kk + hi + 16);
  return cat8(lo, hh);
}

// ---------------- TDM: 2-D tile (nrows x tilecols bf16) global -> LDS ----------------
// D# per CDNA5 ISA ch.8: group0 {count=1, lds_addr, global_addr, type=2},
// group1 {data_size=1(2B), tensor_dim0=rowlen, tensor_dim1=nrows,
//         tile_dim0=tilecols, tile_dim1=nrows, tensor_dim0_stride=rowlen}
static __device__ __forceinline__ void tdm_load_tile(const bf16* gptr, int rowlen,
                                                     int nrows, int tilecols,
                                                     unsigned lds_addr) {
  unsigned long long ga = (unsigned long long)(uintptr_t)gptr;
  v4u g0;
  g0[0] = 1u;                                   // count=1, no gather
  g0[1] = lds_addr;                             // LDS byte address
  g0[2] = (unsigned)(ga & 0xffffffffu);         // global_addr[31:0]
  g0[3] = (unsigned)((ga >> 32) & 0x01ffffffu) | (2u << 30);  // addr[56:32] | type=2
  unsigned td0 = (unsigned)rowlen;
  unsigned td1 = (unsigned)nrows;
  v8i g1;
  g1[0] = (int)(1u << 16);                              // data_size = 2 bytes
  g1[1] = (int)((td0 & 0xffffu) << 16);                 // tensor_dim0[15:0]
  g1[2] = (int)((td0 >> 16) | ((td1 & 0xffffu) << 16)); // tensor_dim0[31:16] | tensor_dim1[15:0]
  g1[3] = (int)((td1 >> 16) | ((unsigned)tilecols << 16)); // tensor_dim1[31:16] | tile_dim0
  g1[4] = (int)td1;                                     // tile_dim1 = nrows, tile_dim2 = 0
  g1[5] = (int)td0;                                     // tensor_dim0_stride[31:0] = rowlen
  g1[6] = 0;
  g1[7] = 0;
  v4i z4;
  z4[0] = 0; z4[1] = 0; z4[2] = 0; z4[3] = 0;
  v8i z8;
#pragma unroll
  for (int i = 0; i < 8; i++) z8[i] = 0;
  __builtin_amdgcn_tensor_load_to_lds(g0, g1, z4, z4, z8, 0);
}

// ---------------- small elementwise / reduction kernels ----------------
__global__ void cvt_f32_bf16(const float* __restrict__ s, bf16* __restrict__ d, int n) {
  int i = blockIdx.x * 256 + threadIdx.x;
  if (i < n) d[i] = (bf16)s[i];
}

__global__ void rms_kernel(const float* __restrict__ x, const float* __restrict__ w,
                           float* __restrict__ of, bf16* __restrict__ ob) {
  int n = blockIdx.x, tid = threadIdx.x;
  const float* row = x + (size_t)n * C_;
  float ss = 0.f;
  for (int c = tid; c < C_; c += 256) { float v = row[c]; ss += v * v; }
  __shared__ float red[8];
  __shared__ float tot;
  for (int off = 16; off; off >>= 1) ss += __shfl_xor(ss, off, 32);
  int wv = tid >> 5, ln = tid & 31;
  if (ln == 0) red[wv] = ss;
  __syncthreads();
  if (wv == 0) {
    float t = (ln < 8) ? red[ln] : 0.f;
    for (int off = 4; off; off >>= 1) t += __shfl_xor(t, off, 32);
    if (ln == 0) tot = t;
  }
  __syncthreads();
  float sc = rsqrtf(tot * (1.f / C_) + 1e-6f);
  for (int c = tid; c < C_; c += 256) {
    float v = row[c] * sc * w[c];
    if (of) of[(size_t)n * C_ + c] = v;
    if (ob) ob[(size_t)n * C_ + c] = (bf16)v;
  }
}

__global__ void conv_kernel(const float* __restrict__ h, const float* __restrict__ cw,
                            bf16* __restrict__ hc) {
  int i = blockIdx.x * 256 + threadIdx.x;
  if (i >= N_ * C_) return;
  int c = i & (C_ - 1);
  int n = i >> 10;
  int t = n & (T_ - 1);
  float w0 = cw[c * 3 + 0], w1 = cw[c * 3 + 1], w2 = cw[c * 3 + 2];
  float v = h[i] * w2;
  if (t >= 1) v += h[i - C_] * w1;
  if (t >= 2) v += h[i - 2 * C_] * w0;
  hc[i] = (bf16)v;
}

// ---------- TDM-staged double-buffered WMMA GEMM (block tile 128x128, Kstep 64) ----------
// Out[M,Nout] = A[M,K] @ W[Nout,K]^T, with epilogues:
//   comb != null: Out[n,c] += comb[n*E+e] * acc   (MoE eo path)
//   else:         Out = acc (+ resid)
__global__ void __launch_bounds__(256) gemm_tdm_kernel(
    const bf16* __restrict__ A, const bf16* __restrict__ W,
    float* __restrict__ Out, const float* __restrict__ resid,
    const float* __restrict__ comb, int e,
    int M, int Nout, int K) {
  __shared__ bf16 Ab[2][128 * 64];   // 16 KB each
  __shared__ bf16 Bb[2][128 * 64];
  int tid = threadIdx.x;
  int w = tid >> 5, lane = tid & 31;
  int lr = lane & 15, hl = lane >> 4;
  int wm = w & 3, wn = w >> 2;
  int mrow0 = blockIdx.x * 128;
  int ncol0 = blockIdx.y * 128;
  int nsteps = K >> 6;

  if (w == 0) {
    tdm_load_tile(A + (size_t)mrow0 * K, K, 128, 64, (unsigned)(uintptr_t)&Ab[0][0]);
    tdm_load_tile(W + (size_t)ncol0 * K, K, 128, 64, (unsigned)(uintptr_t)&Bb[0][0]);
  }
  v8f acc[2][4];
#pragma unroll
  for (int mt = 0; mt < 2; mt++)
#pragma unroll
    for (int nt = 0; nt < 4; nt++) acc[mt][nt] = zero8();

  for (int ks = 0; ks < nsteps; ks++) {
    int cur = ks & 1;
    if (w == 0) {
      if (ks + 1 < nsteps) {
        int nxt = cur ^ 1;
        tdm_load_tile(A + (size_t)mrow0 * K + (ks + 1) * 64, K, 128, 64,
                      (unsigned)(uintptr_t)&Ab[nxt][0]);
        tdm_load_tile(W + (size_t)ncol0 * K + (ks + 1) * 64, K, 128, 64,
                      (unsigned)(uintptr_t)&Bb[nxt][0]);
        __builtin_amdgcn_s_wait_tensorcnt(2);  // current pair done, prefetch in flight
      } else {
        __builtin_amdgcn_s_wait_tensorcnt(0);
      }
    }
    __syncthreads();
    const bf16* abase = &Ab[cur][0];
    const bf16* bbase = &Bb[cur][0];
#pragma unroll
    for (int kk = 0; kk < 64; kk += 32) {
      v16bf afr[2], bfr[4];
#pragma unroll
      for (int mt = 0; mt < 2; mt++)
        afr[mt] = load_a_frag(abase + (wm * 32 + mt * 16 + lr) * 64, kk, lane);
#pragma unroll
      for (int nt = 0; nt < 4; nt++)
        bfr[nt] = load_b_frag(bbase + (wn * 64 + nt * 16 + lr) * 64, kk, lane);
#pragma unroll
      for (int mt = 0; mt < 2; mt++)
#pragma unroll
        for (int nt = 0; nt < 4; nt++)
          acc[mt][nt] = wmma_bf(afr[mt], bfr[nt], acc[mt][nt]);
    }
    asm volatile("s_wait_dscnt 0x0" ::: "memory");
    __syncthreads();
  }

  int colb = ncol0 + wn * 64 + lr;
#pragma unroll
  for (int mt = 0; mt < 2; mt++) {
#pragma unroll
    for (int r = 0; r < 8; r++) {
      int row = mrow0 + wm * 32 + mt * 16 + hl * 8 + r;
      if (comb) {
        float wc = comb[(size_t)row * E_ + e];
#pragma unroll
        for (int nt = 0; nt < 4; nt++) {
          size_t o = (size_t)row * Nout + colb + nt * 16;
          Out[o] += wc * acc[mt][nt][r];
        }
      } else {
#pragma unroll
        for (int nt = 0; nt < 4; nt++) {
          size_t o = (size_t)row * Nout + colb + nt * 16;
          float v = acc[mt][nt][r];
          if (resid) v += resid[o];
          Out[o] = v;
        }
      }
    }
  }
}

// ---------------- RoPE + headwise layernorm + GQA layouts ----------------
__global__ void qkvpost_kernel(const float* __restrict__ qkv, bf16* __restrict__ qb,
                               bf16* __restrict__ kb, bf16* __restrict__ vTb) {
  int n = blockIdx.x;
  int b = n >> 11;
  int t = n & (T_ - 1);
  int y = blockIdx.y;
  int l = threadIdx.x;
  const float* row = qkv + (size_t)n * CKV_;
  if (y < H_ + KV_) {
    const float* hp = (y < H_) ? (row + y * HD_) : (row + C_ + (y - H_) * HD_);
    float x0 = hp[l], x1 = hp[l + 32];
    float inv = __powf(50000.f, -((float)(2 * l)) * (1.f / 64.f));
    float fr = (float)t * inv;
    float cs = __cosf(fr), sn = __sinf(fr);
    float r0 = x0 * cs - x1 * sn;
    float r1 = x1 * cs + x0 * sn;
    float s1 = r0 + r1, s2 = r0 * r0 + r1 * r1;
    for (int off = 16; off; off >>= 1) { s1 += __shfl_xor(s1, off, 32); s2 += __shfl_xor(s2, off, 32); }
    float mean = s1 * (1.f / 64.f);
    float var = s2 * (1.f / 64.f) - mean * mean;
    float rs = rsqrtf(var + 1e-5f);
    float o0 = (r0 - mean) * rs, o1 = (r1 - mean) * rs;
    if (y < H_) {
      bf16* dst = qb + (((size_t)(b * H_ + y) * T_ + t) * HD_);
      dst[l] = (bf16)o0; dst[l + 32] = (bf16)o1;
    } else {
      int kv = y - H_;
      bf16* dst = kb + (((size_t)(b * KV_ + kv) * T_ + t) * HD_);
      dst[l] = (bf16)o0; dst[l + 32] = (bf16)o1;
    }
  } else {
    int kv = y - H_ - KV_;
    const float* hp = row + C_ + KV_ * HD_ + kv * HD_;
    bf16* dst = vTb + ((size_t)(b * KV_ + kv) * HD_) * T_;
    dst[(size_t)l * T_ + t] = (bf16)hp[l];
    dst[(size_t)(l + 32) * T_ + t] = (bf16)hp[l + 32];
  }
}

// ---------------- flash-style causal attention, WMMA QK^T and PV ----------------
__global__ void __launch_bounds__(128) attn_kernel(
    const bf16* __restrict__ qb, const bf16* __restrict__ kb,
    const bf16* __restrict__ vTb, bf16* __restrict__ yb) {
  __shared__ bf16 Pbuf[4][16 * 32];
  int lane = threadIdx.x & 31, wv = threadIdx.x >> 5;
  int b = blockIdx.z, h = blockIdx.y;
  int qt0 = blockIdx.x * 64 + wv * 16;
  int lr = lane & 15, hl = lane >> 4;
  const bf16* qbase = qb + ((size_t)(b * H_ + h) * T_) * HD_;
  const bf16* kbase = kb + ((size_t)(b * KV_ + (h >> 2)) * T_) * HD_;
  const bf16* vbase = vTb + ((size_t)(b * KV_ + (h >> 2)) * HD_) * T_;
  const bf16* qrow = qbase + (size_t)(qt0 + lr) * HD_;
  v16bf qf0 = load_a_frag(qrow, 0, lane);
  v16bf qf1 = load_a_frag(qrow, 32, lane);
  float m[8], ls[8];
  v8f O[4];
#pragma unroll
  for (int r = 0; r < 8; r++) { m[r] = -3.0e38f; ls[r] = 0.f; }
#pragma unroll
  for (int d = 0; d < 4; d++) O[d] = zero8();
  bf16* P = Pbuf[wv];
  for (int kt0 = 0; kt0 < qt0 + 16; kt0 += 32) {
    const bf16* krow0 = kbase + (size_t)(kt0 + lr) * HD_;
    const bf16* krow1 = kbase + (size_t)(kt0 + 16 + lr) * HD_;
    v8f S0 = zero8(), S1 = zero8();
    S0 = wmma_bf(qf0, load_b_frag(krow0, 0, lane), S0);
    S0 = wmma_bf(qf1, load_b_frag(krow0, 32, lane), S0);
    S1 = wmma_bf(qf0, load_b_frag(krow1, 0, lane), S1);
    S1 = wmma_bf(qf1, load_b_frag(krow1, 32, lane), S1);
#pragma unroll
    for (int r = 0; r < 8; r++) {
      int qg = qt0 + r + 8 * hl;
      float v0 = (kt0 + lr      <= qg) ? S0[r] * 0.125f : -3.0e38f;
      float v1 = (kt0 + 16 + lr <= qg) ? S1[r] * 0.125f : -3.0e38f;
      float mx = fmaxf(v0, v1);
      for (int off = 1; off < 16; off <<= 1) mx = fmaxf(mx, __shfl_xor(mx, off, 32));
      float mn = fmaxf(m[r], mx);
      float sc = __expf(m[r] - mn);
      m[r] = mn;
      float p0 = __expf(v0 - mn);
      float p1 = __expf(v1 - mn);
      float rs = p0 + p1;
      for (int off = 1; off < 16; off <<= 1) rs += __shfl_xor(rs, off, 32);
      ls[r] = ls[r] * sc + rs;
#pragma unroll
      for (int d = 0; d < 4; d++) O[d][r] *= sc;
      P[(r + 8 * hl) * 32 + lr] = (bf16)p0;
      P[(r + 8 * hl) * 32 + 16 + lr] = (bf16)p1;
    }
    asm volatile("s_wait_dscnt 0x0" ::: "memory");
    const bf16* prow = P + lr * 32;
    int phi = hl ? 8 : 0;
    v8bf plo = *(const v8bf*)(prow + phi);
    v8bf phh = *(const v8bf*)(prow + 16 + phi);
    v16bf pf = cat8(plo, phh);
#pragma unroll
    for (int d = 0; d < 4; d++) {
      const bf16* vrow = vbase + (size_t)(d * 16 + lr) * T_ + kt0;
      O[d] = wmma_bf(pf, load_b_frag(vrow, 0, lane), O[d]);
    }
    asm volatile("" ::: "memory");
  }
#pragma unroll
  for (int r = 0; r < 8; r++) {
    float inv = 1.f / ls[r];
    int qg = qt0 + r + 8 * hl;
    bf16* dst = yb + (((size_t)b * T_ + qg) * H_ + h) * HD_;
#pragma unroll
    for (int d = 0; d < 4; d++) dst[d * 16 + lr] = (bf16)(O[d][r] * inv);
  }
}

// ---------------- router: logits, softmax, top-2, combine ----------------
__global__ void router_kernel(const bf16* __restrict__ h2, const float* __restrict__ rw,
                              float* __restrict__ probs, float* __restrict__ comb) {
  int n = blockIdx.x, l = threadIdx.x;
  const bf16* hrow = h2 + (size_t)n * C_;
  float lg[E_];
#pragma unroll
  for (int e = 0; e < E_; e++) {
    const float* wrow = rw + e * C_;
    float s = 0.f;
    for (int c = l; c < C_; c += 32) s += (float)hrow[c] * wrow[c];
    for (int off = 16; off; off >>= 1) s += __shfl_xor(s, off, 32);
    lg[e] = s;
  }
  float mx = lg[0];
#pragma unroll
  for (int e = 1; e < E_; e++) mx = fmaxf(mx, lg[e]);
  float p[E_], ssum = 0.f;
#pragma unroll
  for (int e = 0; e < E_; e++) { p[e] = __expf(lg[e] - mx); ssum += p[e]; }
#pragma unroll
  for (int e = 0; e < E_; e++) p[e] /= ssum;
  int i0 = 0;
#pragma unroll
  for (int e = 1; e < E_; e++) if (p[e] > p[i0]) i0 = e;
  int i1 = (i0 == 0) ? 1 : 0;
#pragma unroll
  for (int e = 0; e < E_; e++) if (e != i0 && p[e] > p[i1]) i1 = e;
  float tt = p[i0] + p[i1];
  if (l == 0) {
#pragma unroll
    for (int e = 0; e < E_; e++) {
      probs[(size_t)n * E_ + e] = p[e];
      comb[(size_t)n * E_ + e] = (e == i0) ? p[i0] / tt : ((e == i1) ? p[i1] / tt : 0.f);
    }
  }
}

__global__ void avg_kernel(const float* __restrict__ probs, float* __restrict__ avg) {
  __shared__ float sacc[E_];
  int tid = threadIdx.x;
  if (tid < E_) sacc[tid] = 0.f;
  __syncthreads();
  int n = blockIdx.x * 256 + tid;
  if (n < N_)
    for (int e = 0; e < E_; e++) atomicAdd(&sacc[e], probs[(size_t)n * E_ + e]);
  __syncthreads();
  if (tid < E_) atomicAdd(&avg[tid], sacc[tid]);
}

__global__ void aux_kernel(const float* __restrict__ avg, float* __restrict__ out) {
  float s = 0.f;
  for (int e = 0; e < E_; e++) { float a = avg[e] * (1.f / N_); s += a * a; }
  out[0] = (float)E_ * s;
}

__global__ void silu_kernel(const float* __restrict__ gv, bf16* __restrict__ eh) {
  int i = blockIdx.x * 256 + threadIdx.x;
  if (i >= N_ * F_) return;
  int n = i >> 10, f = i & 1023;
  float g = gv[(size_t)n * (2 * F_) + f];
  float v = gv[(size_t)n * (2 * F_) + F_ + f];
  float sg = g / (1.f + __expf(-g));
  eh[i] = (bf16)(sg * v);
}

// ---------------- host-side orchestration ----------------
extern "C" void kernel_launch(void* const* d_in, const int* in_sizes, int n_in,
                              void* d_out, int out_size, void* d_ws, size_t ws_size,
                              hipStream_t stream) {
  (void)in_sizes; (void)n_in; (void)out_size; (void)ws_size;
  const float* x        = (const float*)d_in[0];
  const float* ln1_w    = (const float*)d_in[1];
  const float* ln2_w    = (const float*)d_in[2];
  const float* conv_w   = (const float*)d_in[3];
  const float* cattn_w  = (const float*)d_in[4];
  const float* cproj_w  = (const float*)d_in[5];
  const float* router_w = (const float*)d_in[6];
  const float* ewv      = (const float*)d_in[7];
  const float* ewo      = (const float*)d_in[8];
  float* out = (float*)d_out;

  char* ws = (char*)d_ws;
  size_t off = 0;
  auto alloc = [&](size_t bytes) -> char* {
    char* p = ws + off;
    off = (off + bytes + 255) & ~(size_t)255;
    return p;
  };
  float* h_f   = (float*)alloc((size_t)N_ * C_ * 4);
  bf16*  hc_b  = (bf16*) alloc((size_t)N_ * C_ * 2);
  bf16*  wattn = (bf16*) alloc((size_t)CKV_ * C_ * 2);
  float* qkv_f = (float*)alloc((size_t)N_ * CKV_ * 4);
  bf16*  q_b   = (bf16*) alloc((size_t)N_ * C_ * 2);
  bf16*  k_b   = (bf16*) alloc((size_t)B_ * KV_ * T_ * HD_ * 2);
  bf16*  vT_b  = (bf16*) alloc((size_t)B_ * KV_ * T_ * HD_ * 2);
  bf16*  y_b   = (bf16*) alloc((size_t)N_ * C_ * 2);
  bf16*  wproj = (bf16*) alloc((size_t)C_ * C_ * 2);
  bf16*  h2_b  = (bf16*) alloc((size_t)N_ * C_ * 2);
  float* probs = (float*)alloc((size_t)N_ * E_ * 4);
  float* comb  = (float*)alloc((size_t)N_ * E_ * 4);
  float* avg   = (float*)alloc(256);
  bf16*  wwv   = (bf16*) alloc((size_t)E_ * 2 * F_ * C_ * 2);
  bf16*  wwo   = (bf16*) alloc((size_t)E_ * C_ * F_ * 2);
  float* gv_f  = (float*)alloc((size_t)N_ * 2 * F_ * 4);
  bf16*  eh_b  = (bf16*) alloc((size_t)N_ * F_ * 2);

  // weight down-conversion to bf16
  cvt_f32_bf16<<<(CKV_ * C_ + 255) / 256, 256, 0, stream>>>(cattn_w, wattn, CKV_ * C_);
  cvt_f32_bf16<<<(C_ * C_ + 255) / 256, 256, 0, stream>>>(cproj_w, wproj, C_ * C_);
  cvt_f32_bf16<<<(E_ * 2 * F_ * C_ + 255) / 256, 256, 0, stream>>>(ewv, wwv, E_ * 2 * F_ * C_);
  cvt_f32_bf16<<<(E_ * C_ * F_ + 255) / 256, 256, 0, stream>>>(ewo, wwo, E_ * C_ * F_);

  // attention branch
  rms_kernel<<<N_, 256, 0, stream>>>(x, ln1_w, h_f, (bf16*)nullptr);
  conv_kernel<<<(N_ * C_ + 255) / 256, 256, 0, stream>>>(h_f, conv_w, hc_b);
  { dim3 g(N_ / 128, CKV_ / 128);
    gemm_tdm_kernel<<<g, 256, 0, stream>>>(hc_b, wattn, qkv_f, nullptr, nullptr, 0,
                                           N_, CKV_, C_); }
  { dim3 g(N_, H_ + 2 * KV_);
    qkvpost_kernel<<<g, 32, 0, stream>>>(qkv_f, q_b, k_b, vT_b); }
  { dim3 g(T_ / 64, H_, B_);
    attn_kernel<<<g, 128, 0, stream>>>(q_b, k_b, vT_b, y_b); }
  { dim3 g(N_ / 128, C_ / 128);
    gemm_tdm_kernel<<<g, 256, 0, stream>>>(y_b, wproj, out, x, nullptr, 0,
                                           N_, C_, C_); }

  // MoE branch
  rms_kernel<<<N_, 256, 0, stream>>>(out, ln2_w, (float*)nullptr, h2_b);
  (void)hipMemsetAsync(avg, 0, E_ * sizeof(float), stream);
  router_kernel<<<N_, 32, 0, stream>>>(h2_b, router_w, probs, comb);
  avg_kernel<<<(N_ + 255) / 256, 256, 0, stream>>>(probs, avg);
  aux_kernel<<<1, 1, 0, stream>>>(avg, out + (size_t)N_ * C_);
  for (int e = 0; e < E_; e++) {
    dim3 gg(N_ / 128, (2 * F_) / 128);
    gemm_tdm_kernel<<<gg, 256, 0, stream>>>(h2_b, wwv + (size_t)e * 2 * F_ * C_,
                                            gv_f, nullptr, nullptr, 0, N_, 2 * F_, C_);
    silu_kernel<<<(N_ * F_ + 255) / 256, 256, 0, stream>>>(gv_f, eh_b);
    dim3 ge(N_ / 128, C_ / 128);
    gemm_tdm_kernel<<<ge, 256, 0, stream>>>(eh_b, wwo + (size_t)e * C_ * F_,
                                            out, nullptr, comb, e, N_, C_, F_);
  }
}
